// BICEPNeuralLayer_47502338294456
// MI455X (gfx1250) — matrix operand order repaired and checked by reading, not compiled, not run on hardware
//
#include <hip/hip_runtime.h>
#include <hip/hip_bf16.h>
#include <math.h>

// ---------------------------------------------------------------------------
// Sizes from the reference
#define B_   256
#define IN_  1024
#define OUT_ 512
#define P_   1000
#define NS_  128
#define SEG  1024                // P padded to 1024 per K-segment
#define K3   (3*SEG)             // 3072: K of GEMM1 ([T | -pLast | -p0])

typedef _Float16 half_t;
typedef __attribute__((ext_vector_type(16))) _Float16 v16h;
typedef __attribute__((ext_vector_type(8)))  _Float16 v8h;
typedef __attribute__((ext_vector_type(8)))  float    v8f;
typedef __attribute__((ext_vector_type(4)))  float    v4f;

__device__ __forceinline__ float sigmoidf_(float z) {
    return 1.0f / (1.0f + __expf(-z));
}

// ---------------------------------------------------------------------------
// K0: feedback[b] = sigmoid(x[b,:] . fb_w + fb_b)      (256 blocks x 256 thr)
__global__ void feedback_kernel(const float* __restrict__ x,
                                const float* __restrict__ fb_w,
                                const float* __restrict__ fb_b,
                                float* __restrict__ fb) {
    int b = blockIdx.x, tid = threadIdx.x;
    __shared__ float red[256];
    float s = 0.0f;
    for (int j = tid; j < IN_; j += 256)
        s += x[(size_t)b * IN_ + j] * fb_w[j];
    red[tid] = s;
    __syncthreads();
    for (int off = 128; off > 0; off >>= 1) {
        if (tid < off) red[tid] += red[tid + off];
        __syncthreads();
    }
    if (tid == 0) fb[b] = sigmoidf_(red[0] + fb_b[0]);
}

// ---------------------------------------------------------------------------
// K1: streaming reduction of noise over NS with analytic weights.
//   wL[s] = retain^(NS-1-s)
//   wT[s] = (1 - retain^(NS-s)) / (decay*dt)  (geometric tail sum)
// Emits X[b, 0*SEG+i] =  c*T,  X[b, SEG+i] = -c*pLast,  X[b, 2*SEG+i] = -c*p0
// where c = fb[b]*sqrt(dt)/NS (folds the conv's time-mean).
// One block per batch row; 250 active lanes each own a float4 (16B) column
// slice -> each time step is one contiguous 4 KB row read via b128 loads.
// This kernel carries the 131 MB HBM stream that dominates runtime, so it
// uses the widest vector loads and unrolls so multiple b128s stay in flight.
__global__ __launch_bounds__(256)
void sde_reduce_kernel(const float* __restrict__ noise,
                       const float* __restrict__ fb,
                       const float* __restrict__ decay_param,
                       half_t* __restrict__ X) {
    int b = blockIdx.x;
    int t = threadIdx.x;                          // 0..255; t<250 active
    float decay  = sigmoidf_(decay_param[0]) * 0.5f;
    float dt     = 1.0f / (float)NS_;
    float retain = 1.0f - decay * dt;
    float invd   = 1.0f / (decay * dt);
    float c      = fb[b] * sqrtf(dt) * (1.0f / (float)NS_);

    v4f sT = {0.f, 0.f, 0.f, 0.f};
    v4f sL = {0.f, 0.f, 0.f, 0.f};
    v4f s0 = {0.f, 0.f, 0.f, 0.f};
    if (t < P_ / 4) {
        const v4f* np = (const v4f*)(noise + (size_t)b * NS_ * P_) + t;
        float wl = 1.0f;                          // retain^(NS-1-s), s descending
#pragma unroll 4
        for (int s = NS_ - 1; s >= 0; --s) {
            v4f v  = np[(size_t)s * (P_ / 4)];
            float wt = (1.0f - retain * wl) * invd;
            sT += wt * v;
            sL += wl * v;
            if (s == 0) s0 = v;
            wl *= retain;
        }
    }
    half_t* xr = X + (size_t)b * K3;
    int i = 4 * t;                                // 0..1020 (>=1000 -> zeros)
#pragma unroll
    for (int j = 0; j < 4; ++j) {
        xr[i + j]           = (half_t)( c * sT[j]);
        xr[SEG + i + j]     = (half_t)(-c * sL[j]);
        xr[2 * SEG + i + j] = (half_t)(-c * s0[j]);
    }
}

// ---------------------------------------------------------------------------
// K2: Wcat[p, :] = [ W0+W1+W2 | W0 | W2 ] in f16, zero-padded to 1024x3072.
__global__ void build_wcat_kernel(const float* __restrict__ conv_w,
                                  half_t* __restrict__ Wcat) {
    int p = blockIdx.x;                            // 0..1023
    half_t* wr = Wcat + (size_t)p * K3;
    for (int j = 0; j < 4; ++j) {
        int i = threadIdx.x + j * 256;             // 0..1023
        float w0 = 0.f, w1 = 0.f, w2 = 0.f;
        if (p < P_ && i < P_) {
            const float* cp = conv_w + ((size_t)p * P_ + i) * 3;
            w0 = cp[0]; w1 = cp[1]; w2 = cp[2];
        }
        wr[i]           = (half_t)(w0 + w1 + w2);
        wr[SEG + i]     = (half_t)w0;
        wr[2 * SEG + i] = (half_t)w2;
    }
}

// K3: out_w -> f16, K padded 1000->1024
__global__ void build_outw_kernel(const float* __restrict__ out_w,
                                  half_t* __restrict__ W) {
    int o = blockIdx.x;                            // 0..511
    for (int j = 0; j < 4; ++j) {
        int k = threadIdx.x + j * 256;
        float v = (k < P_) ? out_w[(size_t)o * P_ + k] : 0.0f;
        W[(size_t)o * SEG + k] = (half_t)v;
    }
}

// ---------------------------------------------------------------------------
// WMMA GEMM: D[M,N] = A[M,K] * Bm[N,K]^T + bias[n]
// 128 threads (4 waves), 64x64 tile per block, each wave owns a 32x32 quadrant
// (2x2 of 16x16 WMMA tiles).  Fragments loaded straight from global (operands
// are L2-resident).  Lane layouts used here (contiguous K, same for A and B):
//   A 16x32 f16 : lane l holds M=l%16; half h -> K = k0+(l/16)*16+h
//   B 32x16 f16 : lane l holds N=l%16; half h -> K = k0+(l/16)*16+h
//   C/D 16x16 f32: lane l, reg r -> (M=(l/16)*8+r, N=l%16)
template <int KDIM, bool OUT_HALF>
__global__ __launch_bounds__(128)
void gemm_wmma_kernel(const half_t* __restrict__ A,
                      const half_t* __restrict__ Bm,
                      const float* __restrict__ bias, int biasN,
                      void* __restrict__ Out, int ldOut) {
    const int lane = threadIdx.x & 31;
    const int wave = threadIdx.x >> 5;
    const int lrow = lane & 15;        // l % 16
    const int lhi  = lane >> 4;        // l / 16
    const int wm   = (wave >> 1) * 2;  // 0 or 2
    const int wn   = (wave & 1) * 2;   // 0 or 2
    const int m0   = blockIdx.y * 64;
    const int n0   = blockIdx.x * 64;

    v8f c[2][2] = {};

    for (int k0 = 0; k0 < KDIM; k0 += 32) {
        v16h a[2], bfrag[2];
#pragma unroll
        for (int mi = 0; mi < 2; ++mi) {
            const half_t* ap = A + (size_t)(m0 + (wm + mi) * 16 + lrow) * KDIM
                                 + k0 + lhi * 16;
            v8h lo = *(const v8h*)ap;
            v8h hi = *(const v8h*)(ap + 8);
            a[mi] = __builtin_shufflevector(lo, hi, 0, 1, 2, 3, 4, 5, 6, 7,
                                            8, 9, 10, 11, 12, 13, 14, 15);
        }
#pragma unroll
        for (int ni = 0; ni < 2; ++ni) {
            const half_t* bp = Bm + (size_t)(n0 + (wn + ni) * 16 + lrow) * KDIM
                                  + k0 + lhi * 16;
            bfrag[ni] = *(const v16h*)bp;
        }
#pragma unroll
        for (int mi = 0; mi < 2; ++mi)
#pragma unroll
            for (int ni = 0; ni < 2; ++ni)
                c[mi][ni] = __builtin_amdgcn_wmma_f32_16x16x32_f16(
                    false, a[mi], false, bfrag[ni], (short)0, c[mi][ni],
                    false, false);
    }

#pragma unroll
    for (int mi = 0; mi < 2; ++mi) {
#pragma unroll
        for (int ni = 0; ni < 2; ++ni) {
            int n  = n0 + (wn + ni) * 16 + lrow;
            float bv = (bias != nullptr && n < biasN) ? bias[n] : 0.0f;
#pragma unroll
            for (int r = 0; r < 8; ++r) {
                int m = m0 + (wm + mi) * 16 + lhi * 8 + r;
                float v = c[mi][ni][r] + bv;
                if (OUT_HALF)
                    ((half_t*)Out)[(size_t)m * ldOut + n] = (half_t)v;
                else
                    ((float*)Out)[(size_t)m * ldOut + n] = v;
            }
        }
    }
}

// ---------------------------------------------------------------------------
extern "C" void kernel_launch(void* const* d_in, const int* in_sizes, int n_in,
                              void* d_out, int out_size, void* d_ws, size_t ws_size,
                              hipStream_t stream) {
    (void)in_sizes; (void)n_in; (void)out_size; (void)ws_size;
    const float* x           = (const float*)d_in[0];   // [256,1024]
    const float* noise       = (const float*)d_in[1];   // [256,128,1000]
    const float* fb_w        = (const float*)d_in[2];   // [1,1024]
    const float* fb_b        = (const float*)d_in[3];   // [1]
    const float* decay_param = (const float*)d_in[4];   // [1]
    const float* conv_w      = (const float*)d_in[5];   // [1000,1000,3]
    const float* conv_b      = (const float*)d_in[6];   // [1000]
    const float* out_w       = (const float*)d_in[7];   // [512,1000]
    const float* out_b       = (const float*)d_in[8];   // [512]
    float* out               = (float*)d_out;           // [256,512]

    // Workspace carve-out (256B aligned), ~9.4 MB total.
    char* ws = (char*)d_ws;
    size_t off = 0;
    auto carve = [&](size_t bytes) {
        char* p = ws + off;
        off = (off + bytes + 255) & ~(size_t)255;
        return p;
    };
    float*  fb     = (float*)carve((size_t)B_ * sizeof(float));
    half_t* X      = (half_t*)carve((size_t)B_ * K3 * sizeof(half_t));
    half_t* Wcat   = (half_t*)carve((size_t)SEG * K3 * sizeof(half_t));
    half_t* agg16  = (half_t*)carve((size_t)B_ * SEG * sizeof(half_t));
    half_t* outw16 = (half_t*)carve((size_t)OUT_ * SEG * sizeof(half_t));

    feedback_kernel<<<B_, 256, 0, stream>>>(x, fb_w, fb_b, fb);
    sde_reduce_kernel<<<B_, 256, 0, stream>>>(noise, fb, decay_param, X);
    build_wcat_kernel<<<SEG, 256, 0, stream>>>(conv_w, Wcat);
    build_outw_kernel<<<OUT_, 256, 0, stream>>>(out_w, outw16);

    // GEMM1: agg16[256,1024] = X[256,3072] * Wcat[1024,3072]^T + conv_b
    gemm_wmma_kernel<K3, true><<<dim3(SEG / 64, B_ / 64), 128, 0, stream>>>(
        X, Wcat, conv_b, P_, agg16, SEG);
    // GEMM2: out[256,512] = agg16[256,1024] * outw16[512,1024]^T + out_b
    gemm_wmma_kernel<SEG, false><<<dim3(OUT_ / 64, B_ / 64), 128, 0, stream>>>(
        agg16, outw16, out_b, OUT_, out, OUT_);
}
